// RecursiveIntegrator_19894288515167
// MI455X (gfx1250) — matrix operand order repaired and checked by reading, not compiled, not run on hardware
//
#include <hip/hip_runtime.h>
#include <math.h>

#define D_MODEL 1024
#define BATCH   8
#define SEQ     512
#define BS      (BATCH * SEQ)      // 4096 rows
#define D2      (D_MODEL / 2)      // 512
#define TD      (3 * D_MODEL)      // 3072
#define DEPTH   3
#define EPS_RMS 1.1920929e-07f

typedef __attribute__((ext_vector_type(16))) __bf16        v16bf;
typedef __attribute__((ext_vector_type(8)))  float          v8f;
typedef __attribute__((ext_vector_type(8)))  unsigned short u16x8;
typedef __attribute__((ext_vector_type(4)))  unsigned int   u32x4;
typedef __attribute__((ext_vector_type(8)))  int            i32x8;
typedef __attribute__((ext_vector_type(4)))  int            i32x4;

// ---- CDNA5 data-mover feature detection (toolchain-portable) -------------
#if defined(__has_builtin)
#  if __has_builtin(__builtin_amdgcn_tensor_load_to_lds) && \
      __has_builtin(__builtin_amdgcn_s_wait_tensorcnt)
#    if __has_include(<hip/amd_detail/amd_gfx1250_TDM.h>)
#      define TDM_ARITY 6
#    else
#      define TDM_ARITY 5
#    endif
#  endif
#  if __has_builtin(__builtin_amdgcn_global_load_async_to_lds_b128) && \
      __has_builtin(__builtin_amdgcn_s_wait_asynccnt)
#    define HAS_ASYNC_LDS 1
#  endif
#endif

#define AS3(p)  ((__attribute__((address_space(3))) void*)(p))
#define AS1C(p) ((const __attribute__((address_space(1))) void*)(p))

static __device__ __forceinline__ unsigned short f2bf(float f) {
    unsigned u = __builtin_bit_cast(unsigned, f);
    u += 0x7FFFu + ((u >> 16) & 1u);   // round-to-nearest-even
    return (unsigned short)(u >> 16);
}
static __device__ __forceinline__ __bf16 bfu(unsigned short v) {
    return __builtin_bit_cast(__bf16, v);
}

// ---------------------------------------------------------------- utilities
__global__ void k_f32_to_bf16(const float* __restrict__ src,
                              unsigned short* __restrict__ dst, int n) {
    int i = blockIdx.x * blockDim.x + threadIdx.x;
    if (i < n) dst[i] = f2bf(src[i]);
}

__global__ void k_zero(float* __restrict__ p, int n) {
    int i = blockIdx.x * blockDim.x + threadIdx.x;
    if (i < n) p[i] = 0.f;
}

// ------------------------------------------------- WMMA GEMM: C = A*W^T + b
// A: [M,K] fp32 row-major. Wb: [N,K] bf16 row-major. C: [M,N] fp32.
// Block = 256 threads (8 waves), tile 64x64, K-chunk 32.
// LDS row stride = 40 u16 (32 data + 8 pad): 16B-aligned rows, bank spread.
// B tile is staged by the Tensor Data Mover (pad feature writes the same
// 16-data-DWORD + 4-pad-DWORD row pitch), falling back to async-to-LDS
// loads, then to manual copies, depending on toolchain support.
__global__ __launch_bounds__(256) void k_gemm_bf16_wmma(
    const float* __restrict__ A, const unsigned short* __restrict__ Wb,
    const float* __restrict__ bias, float* __restrict__ C,
    int M, int N, int K)
{
    __shared__ unsigned short As[64][40];
    __shared__ unsigned short Bs[64][40];
    const int tid  = threadIdx.x;
    const int lane = tid & 31;
    const int wave = tid >> 5;              // 0..7
    const int m0   = blockIdx.y << 6;
    const int n0   = blockIdx.x << 6;
    const int mi   = wave & 3;              // 16-row subtile 0..3
    const int nb   = (wave >> 2) << 1;      // 16-col subtile base {0,2}
    const int hi   = lane >> 4;             // half-wave
    const int l16  = lane & 15;

    v8f acc0 = {};
    v8f acc1 = {};

    for (int k0 = 0; k0 < K; k0 += 32) {
        // ---- stage A (fp32 -> bf16 conversion through VGPRs) ----
#pragma unroll
        for (int i = 0; i < 8; ++i) {       // 2048 elems, 8 per thread
            int idx = tid + (i << 8);
            int r = idx >> 5, c = idx & 31;
            As[r][c] = f2bf(A[(size_t)(m0 + r) * K + k0 + c]);
        }
        // ---- stage B (bf16 tile, straight global -> LDS) ----
#if defined(TDM_ARITY)
        if (wave == 0) {
            const unsigned long long ga =
                (unsigned long long)(const char*)(Wb + (size_t)n0 * K + k0);
            const unsigned lds_off =
                (unsigned)(unsigned long long)
                ((__attribute__((address_space(3))) unsigned short*)&Bs[0][0]);
            u32x4 g0;
            g0[0] = 1u;                                        // count=1, load
            g0[1] = lds_off;                                   // lds_addr
            g0[2] = (unsigned)ga;                              // global_addr lo
            g0[3] = (unsigned)((ga >> 32) & 0x01FFFFFFu) | (2u << 30); // hi|type=2
            const int td1 = N - n0;                            // rows below tile
            i32x8 g1;
            g1[0] = (1 << 16)      // data_size = 2 bytes
                  | (1 << 20)      // pad_enable
                  | (3 << 22)      // pad_interval: 16 DWORDs of data
                  | (3 << 25);     // pad_amount:   4 DWORDs of pad
            g1[1] = (int)((unsigned)(K & 0xFFFF) << 16);       // tensor_dim0 lo
            g1[2] = (int)(((unsigned)K >> 16) | ((unsigned)(td1 & 0xFFFF) << 16));
            g1[3] = (int)(((unsigned)td1 >> 16) | (32u << 16));// tile_dim0 = 32
            g1[4] = 64;                                        // tile_dim1 = 64
            g1[5] = K;                                         // dim0_stride lo
            g1[6] = 0;
            g1[7] = 0;
            i32x4 gz = {0, 0, 0, 0};
#if TDM_ARITY == 6
            i32x8 gz8 = {0, 0, 0, 0, 0, 0, 0, 0};
            __builtin_amdgcn_tensor_load_to_lds(g0, g1, gz, gz, gz8, 0);
#else
            __builtin_amdgcn_tensor_load_to_lds(g0, g1, gz, gz, 0);
#endif
            __builtin_amdgcn_s_wait_tensorcnt(0);
        }
#elif defined(HAS_ASYNC_LDS)
        {
            int r = tid >> 2, c = (tid & 3) << 3;    // 64 rows x 4 b128 chunks
            __builtin_amdgcn_global_load_async_to_lds_b128(
                AS1C(Wb + (size_t)(n0 + r) * K + k0 + c), AS3(&Bs[r][c]), 0, 0);
            __builtin_amdgcn_s_wait_asynccnt(0);
        }
#else
#pragma unroll
        for (int i = 0; i < 8; ++i) {
            int idx = tid + (i << 8);
            int r = idx >> 5, c = idx & 31;
            Bs[r][c] = Wb[(size_t)(n0 + r) * K + k0 + c];
        }
#endif
        __syncthreads();

        v16bf a, bA, bB;
        const int mrow  = (mi << 4) + l16;
        const int nrow0 = (nb << 4) + l16;
        const int nrow1 = nrow0 + 16;
#pragma unroll
        for (int e = 0; e < 16; ++e) {
            // A fragment K map: K = (e&7) + 16*(e>=8) + 8*hi
            int ka = (e & 7) + ((e >> 3) << 4) + (hi << 3);
            // B fragment K map: K = (e&7) + 16*(e>=8) + 8*hi  (B[k][n] = W[n][k])
            int kb = (e & 7) + ((e >> 3) << 4) + (hi << 3);
            a[e]  = bfu(As[mrow][ka]);
            bA[e] = bfu(Bs[nrow0][kb]);
            bB[e] = bfu(Bs[nrow1][kb]);
        }
        acc0 = __builtin_amdgcn_wmma_f32_16x16x32_bf16(false, a, false, bA,
                                                       (short)0, acc0, false, false);
        acc1 = __builtin_amdgcn_wmma_f32_16x16x32_bf16(false, a, false, bB,
                                                       (short)0, acc1, false, false);
        __syncthreads();
    }

    const int nc0 = n0 + (nb << 4) + l16;
    const int nc1 = nc0 + 16;
    const int mb  = m0 + (mi << 4) + (hi << 3);  // C layout: M = r + 8*hi
    const float bv0 = bias[nc0];
    const float bv1 = bias[nc1];
#pragma unroll
    for (int r = 0; r < 8; ++r) {
        C[(size_t)(mb + r) * N + nc0] = acc0[r] + bv0;
        C[(size_t)(mb + r) * N + nc1] = acc1[r] + bv1;
    }
}

// ------------------------------------------------------- RMSNorm + GELU
__global__ __launch_bounds__(256) void k_rms_gelu(float* __restrict__ io,
                                                  const float* __restrict__ w) {
    __shared__ float red[256];
    float* p = io + (size_t)blockIdx.x * D_MODEL;
    float v[4]; float ss = 0.f;
#pragma unroll
    for (int i = 0; i < 4; ++i) { v[i] = p[threadIdx.x + (i << 8)]; ss += v[i] * v[i]; }
    red[threadIdx.x] = ss; __syncthreads();
    for (int o = 128; o > 0; o >>= 1) {
        if (threadIdx.x < o) red[threadIdx.x] += red[threadIdx.x + o];
        __syncthreads();
    }
    const float scale = rsqrtf(red[0] * (1.f / D_MODEL) + EPS_RMS);
#pragma unroll
    for (int i = 0; i < 4; ++i) {
        int c = threadIdx.x + (i << 8);
        float xn = v[i] * scale * w[c];
        p[c] = 0.5f * xn * (1.f + erff(xn * 0.70710678118654752f));  // exact GELU
    }
}

// --------------------------------------- RMSNorm + weighted residual blend
__global__ __launch_bounds__(256) void k_rms_residual(
    const float* __restrict__ h2, const float* __restrict__ xin,
    const float* __restrict__ w, const float* __restrict__ resw_p,
    float* __restrict__ out)
{
    __shared__ float red[256];
    const size_t base = (size_t)blockIdx.x * D_MODEL;
    float v[4]; float ss = 0.f;
#pragma unroll
    for (int i = 0; i < 4; ++i) { v[i] = h2[base + threadIdx.x + (i << 8)]; ss += v[i] * v[i]; }
    red[threadIdx.x] = ss; __syncthreads();
    for (int o = 128; o > 0; o >>= 1) {
        if (threadIdx.x < o) red[threadIdx.x] += red[threadIdx.x + o];
        __syncthreads();
    }
    const float scale = rsqrtf(red[0] * (1.f / D_MODEL) + EPS_RMS);
    const float rw = resw_p[0];
#pragma unroll
    for (int i = 0; i < 4; ++i) {
        int c = threadIdx.x + (i << 8);
        float hn = v[i] * scale * w[c];
        float xv = xin[base + c];
        out[base + c] = xv + rw * (hn - xv);
    }
}

// -------------------------------------------------- persistent GRU scan
// One 1024-thread workgroup; h lives in LDS; 512 sequential steps.
// gh = h @ Whh^T via WMMA with Whh-tile as A (16 n-rows x 32 k) and h^T as B.
__global__ __launch_bounds__(1024) void k_gru_scan(
    const unsigned short* __restrict__ Whhb,   // [3D, D] bf16
    const float* __restrict__ gi,              // [B*S, 3D] fp32 (incl. bih)
    const float* __restrict__ bhh,             // [3D]
    float* __restrict__ hid,                   // [B, D] in/out
    float* __restrict__ out)                   // [B, S, D]
{
    extern __shared__ char smem[];
    float*          hfp = (float*)smem;                              // 8x1024 f32
    unsigned short* hbf = (unsigned short*)(smem + 32768);           // 8x1024 bf16
    float*          gh  = (float*)(smem + 32768 + 16384);            // 8x3072 f32

    const int tid  = threadIdx.x;
    const int lane = tid & 31;
    const int wave = tid >> 5;          // 0..31
    const int hi   = lane >> 4;
    const int l16  = lane & 15;
    const int bread = l16 & 7;          // clamp batch col for B-frag reads

    for (int i = tid; i < BATCH * D_MODEL; i += 1024) {
        float v = hid[i]; hfp[i] = v; hbf[i] = f2bf(v);
    }
    __syncthreads();

    for (int s = 0; s < SEQ; ++s) {
        // --- recurrent GEMM: 192 n-tiles, 6 per wave ---
#pragma unroll 1
        for (int t = 0; t < 6; ++t) {
            const int nt   = wave * 6 + t;
            const int nrow = (nt << 4) + l16;
            const unsigned short* wrow = Whhb + (size_t)nrow * D_MODEL;
            v8f acc = {};
            for (int kc = 0; kc < D_MODEL; kc += 32) {
                if (kc + 256 <= D_MODEL) __builtin_prefetch(wrow + kc + 256, 0, 1);
                // A (Whh) fragment: two contiguous 8x-bf16 chunks per lane
                u16x8 a_lo = *(const u16x8*)(wrow + kc + (hi << 3));
                u16x8 a_hi = *(const u16x8*)(wrow + kc + 16 + (hi << 3));
                // B (h^T) fragment: N=batch col, two 8x-bf16 K chunks from LDS
                const unsigned short* hrow = hbf + bread * D_MODEL + kc + (hi << 3);
                u16x8 b_lo = *(const u16x8*)(hrow);
                u16x8 b_hi = *(const u16x8*)(hrow + 16);
                v16bf a, b;
#pragma unroll
                for (int e = 0; e < 8; ++e) {
                    a[e] = bfu(a_lo[e]); a[e + 8] = bfu(a_hi[e]);
                    b[e] = bfu(b_lo[e]); b[e + 8] = bfu(b_hi[e]);
                }
                acc = __builtin_amdgcn_wmma_f32_16x16x32_bf16(false, a, false, b,
                                                              (short)0, acc, false, false);
            }
            if (l16 < 8) {     // D layout: col = lane&15 (batch), row = r + 8*hi
#pragma unroll
                for (int r = 0; r < 8; ++r)
                    gh[l16 * TD + (nt << 4) + r + (hi << 3)] = acc[r];
            }
        }
        __syncthreads();

        // --- gate update: 8192 h-elements, 8 per thread ---
        for (int i = tid; i < BATCH * D_MODEL; i += 1024) {
            const int b = i >> 10;
            const int d = i & 1023;
            const float* gir = gi + (size_t)(b * SEQ + s) * TD;
            float hr = gh[b * TD + d]               + bhh[d];
            float hz = gh[b * TD + D_MODEL + d]     + bhh[D_MODEL + d];
            float hg = gh[b * TD + 2 * D_MODEL + d] + bhh[2 * D_MODEL + d];
            float r  = 1.f / (1.f + expf(-(gir[d] + hr)));
            float z  = 1.f / (1.f + expf(-(gir[D_MODEL + d] + hz)));
            float n  = tanhf(gir[2 * D_MODEL + d] + r * hg);
            float hn = (1.f - z) * n + z * hfp[i];
            out[((size_t)b * SEQ + s) * D_MODEL + d] = hn;
            hfp[i] = hn;
            hbf[i] = f2bf(hn);
        }
        __syncthreads();
    }

    for (int i = tid; i < BATCH * D_MODEL; i += 1024) hid[i] = hfp[i];
}

// ------------------------------------------- per-row L2 norm, batch-mean acc
__global__ __launch_bounds__(256) void k_rownorm_acc(
    const float* __restrict__ X, float scale, float* __restrict__ acc)
{
    __shared__ float red[256];
    const float* p = X + (size_t)blockIdx.x * D2;
    float a = p[threadIdx.x], b = p[threadIdx.x + 256];
    red[threadIdx.x] = a * a + b * b;
    __syncthreads();
    for (int o = 128; o > 0; o >>= 1) {
        if (threadIdx.x < o) red[threadIdx.x] += red[threadIdx.x + o];
        __syncthreads();
    }
    if (threadIdx.x == 0)
        atomicAdd(&acc[blockIdx.x / SEQ], sqrtf(red[0]) * scale);
}

__global__ void k_phi(const float* __restrict__ acc, const float* __restrict__ sc,
                      const float* __restrict__ bi, float* __restrict__ out)
{
    int b = threadIdx.x;
    if (b < BATCH) {
        float w = acc[b], p = acc[8 + b];
        float phi = sc[0] * ((w - p) / (w + 1e-8f)) + bi[0];
        out[b] = fminf(fmaxf(phi, 0.f), 1.f);
    }
}

// =========================================================================
extern "C" void kernel_launch(void* const* d_in, const int* in_sizes, int n_in,
                              void* d_out, int out_size, void* d_ws, size_t ws_size,
                              hipStream_t stream)
{
    (void)in_sizes; (void)n_in; (void)out_size; (void)ws_size;
    const float* x_in   = (const float*)d_in[0];
    const float* w1     = (const float*)d_in[1];
    const float* b1     = (const float*)d_in[2];
    const float* rms1   = (const float*)d_in[3];
    const float* w2     = (const float*)d_in[4];
    const float* b2     = (const float*)d_in[5];
    const float* rms2   = (const float*)d_in[6];
    const float* resw   = (const float*)d_in[7];
    const float* wholew = (const float*)d_in[8];
    const float* wholeb = (const float*)d_in[9];
    const float* partsw = (const float*)d_in[10];
    const float* partsb = (const float*)d_in[11];
    const float* phis   = (const float*)d_in[12];
    const float* phib   = (const float*)d_in[13];
    const float* wih[2] = {(const float*)d_in[14], (const float*)d_in[18]};
    const float* whh[2] = {(const float*)d_in[15], (const float*)d_in[19]};
    const float* bih[2] = {(const float*)d_in[16], (const float*)d_in[20]};
    const float* bhh[2] = {(const float*)d_in[17], (const float*)d_in[21]};

    char* ws = (char*)d_ws;
    size_t off = 0;
    auto take = [&](size_t bytes) -> char* {
        char* p = ws + off;
        off += (bytes + 255) & ~(size_t)255;
        return p;
    };
    float* xA    = (float*)take((size_t)BS * D_MODEL * 4);
    float* xB    = (float*)take((size_t)BS * D_MODEL * 4);
    float* tmp1  = (float*)take((size_t)BS * D_MODEL * 4);
    float* tmp2  = (float*)take((size_t)BS * D_MODEL * 4);
    float* hist0 = (float*)take((size_t)BS * D_MODEL * 4);
    float* hist1 = (float*)take((size_t)BS * D_MODEL * 4);
    float* gibuf = (float*)take((size_t)BS * TD * 4);
    float* proj  = (float*)take((size_t)BS * D2 * 4);
    float* hid   = (float*)take((size_t)2 * BATCH * D_MODEL * 4);
    float* acc   = (float*)take(16 * 4);
    unsigned short* w1b  = (unsigned short*)take((size_t)D_MODEL * D_MODEL * 2);
    unsigned short* w2b  = (unsigned short*)take((size_t)D_MODEL * D_MODEL * 2);
    unsigned short* whob = (unsigned short*)take((size_t)D2 * D_MODEL * 2);
    unsigned short* parb = (unsigned short*)take((size_t)D2 * D_MODEL * 2);
    unsigned short* wihb[2], *whhb[2];
    for (int l = 0; l < 2; ++l) {
        wihb[l] = (unsigned short*)take((size_t)TD * D_MODEL * 2);
        whhb[l] = (unsigned short*)take((size_t)TD * D_MODEL * 2);
    }

    auto conv = [&](const float* s, unsigned short* d, size_t n) {
        k_f32_to_bf16<<<dim3((unsigned)((n + 255) / 256)), 256, 0, stream>>>(s, d, (int)n);
    };
    conv(w1, w1b, (size_t)D_MODEL * D_MODEL);
    conv(w2, w2b, (size_t)D_MODEL * D_MODEL);
    conv(wholew, whob, (size_t)D2 * D_MODEL);
    conv(partsw, parb, (size_t)D2 * D_MODEL);
    for (int l = 0; l < 2; ++l) {
        conv(wih[l], wihb[l], (size_t)TD * D_MODEL);
        conv(whh[l], whhb[l], (size_t)TD * D_MODEL);
    }

    k_zero<<<dim3((2 * BATCH * D_MODEL + 255) / 256), 256, 0, stream>>>(hid, 2 * BATCH * D_MODEL);
    k_zero<<<1, 16, 0, stream>>>(acc, 16);
    hipMemcpyAsync(xA, x_in, (size_t)BS * D_MODEL * 4, hipMemcpyDeviceToDevice, stream);

    auto gemm = [&](const float* A, const unsigned short* Wb, const float* bias,
                    float* C, int M, int N, int K) {
        k_gemm_bf16_wmma<<<dim3(N / 64, M / 64), 256, 0, stream>>>(A, Wb, bias, C, M, N, K);
    };

    const size_t GRU_LDS = 32768 + 16384 + (size_t)BATCH * TD * 4;   // 147456 B
    float* xin = xA; float* xout = xB;
    float* hist[2] = {hist0, hist1};

    for (int dep = 0; dep < DEPTH; ++dep) {
        gemm(xin, w1b, b1, tmp1, BS, D_MODEL, D_MODEL);
        k_rms_gelu<<<BS, 256, 0, stream>>>(tmp1, rms1);
        gemm(tmp1, w2b, b2, tmp2, BS, D_MODEL, D_MODEL);
        k_rms_residual<<<BS, 256, 0, stream>>>(tmp2, xin, rms2, resw, tmp1);

        gemm(tmp1, wihb[0], bih[0], gibuf, BS, TD, D_MODEL);
        k_gru_scan<<<1, 1024, GRU_LDS, stream>>>(whhb[0], gibuf, bhh[0], hid, tmp2);
        gemm(tmp2, wihb[1], bih[1], gibuf, BS, TD, D_MODEL);
        k_gru_scan<<<1, 1024, GRU_LDS, stream>>>(whhb[1], gibuf, bhh[1],
                                                 hid + BATCH * D_MODEL, xout);
        if (dep < 2)
            hipMemcpyAsync(hist[dep], xout, (size_t)BS * D_MODEL * 4,
                           hipMemcpyDeviceToDevice, stream);
        float* t = xin; xin = xout; xout = t;
    }

    // Phi only matters at the final depth (earlier values are overwritten).
    gemm(xin, whob, wholeb, proj, BS, D2, D_MODEL);
    k_rownorm_acc<<<BS, 256, 0, stream>>>(proj, 1.f / SEQ, acc);          // whole
    for (int h = 0; h < 2; ++h) {
        gemm(hist[h], parb, partsb, proj, BS, D2, D_MODEL);
        k_rownorm_acc<<<BS, 256, 0, stream>>>(proj, 0.5f / SEQ, acc + 8); // parts
    }
    k_phi<<<1, 32, 0, stream>>>(acc, phis, phib, (float*)d_out + (size_t)BS * D_MODEL);
    hipMemcpyAsync(d_out, xin, (size_t)BS * D_MODEL * 4, hipMemcpyDeviceToDevice, stream);
}
